// RelPartialLearnableMultiHeadAttn_48996986912857
// MI455X (gfx1250) — matrix operand order repaired and checked by reading, not compiled, not run on hardware
//
#include <hip/hip_runtime.h>
#include <hip/hip_bf16.h>

#define QLEN   1024
#define MLEN   1024
#define KLEN   2048
#define BSZ    2
#define DMODEL 1024
#define NHEAD  16
#define DHEAD  64
#define QKV_N  (3 * NHEAD * DHEAD)   // 3072
#define ATT_SCALE 0.125f             // 1/sqrt(64)
#define LN_EPS 1e-5f

typedef __attribute__((ext_vector_type(16))) __bf16 v16bf;
typedef __attribute__((ext_vector_type(8)))  float  v8f;

union ABf {
  v16bf v;
  unsigned int u[8];
  unsigned short s[16];
};

__device__ __forceinline__ unsigned short f32_to_bf16_bits(float f) {
  union { float f; unsigned int u; } cv;
  cv.f = f;
  unsigned int u = cv.u;
  u += 0x7FFFu + ((u >> 16) & 1u);   // round-to-nearest-even
  return (unsigned short)(u >> 16);
}

__device__ __forceinline__ v8f v8f_zero() {
  v8f z = {0.f, 0.f, 0.f, 0.f, 0.f, 0.f, 0.f, 0.f};
  return z;
}

__device__ __forceinline__ v8f wmma_bf16(const ABf& a, const ABf& b, v8f c) {
  return __builtin_amdgcn_wmma_f32_16x16x32_bf16(
      /*neg_a=*/false, a.v, /*neg_b=*/false, b.v,
      /*c_mod=*/(short)0, c, /*reuse_a=*/false, /*reuse_b=*/false);
}

// ---------------------------------------------------------------------------
// CDNA5 async global->LDS staging (ASYNCcnt path), with safe fallback.
// Builtin parameter types per hipcc diagnostic:
//   arg0: int __attribute__((vector_size(16))) __device__*   (AS1, global)
//   arg1: same vector type in AS3 (LDS)
// LDS segment address = low 32 bits of the flat pointer (ISA: LDS aperture
// truncates addr[31:0]).
// ---------------------------------------------------------------------------
#ifndef __has_builtin
#define __has_builtin(x) 0
#endif

#if __has_builtin(__builtin_amdgcn_global_load_async_to_lds_b128)
#define ASYNC_LDS 1
typedef int v4i_async __attribute__((vector_size(16)));
typedef __attribute__((address_space(1))) v4i_async* g1_v4i;
typedef __attribute__((address_space(3))) v4i_async* l3_v4i;
__device__ __forceinline__ void async_b128(const void* g, void* l) {
  __builtin_amdgcn_global_load_async_to_lds_b128(
      (g1_v4i)(unsigned long long)g,
      (l3_v4i)(unsigned int)(unsigned long long)l, 0, 0);
}
#else
#define ASYNC_LDS 0
#endif

__device__ __forceinline__ void wait_async_all() {
#if __has_builtin(__builtin_amdgcn_s_wait_asynccnt)
  __builtin_amdgcn_s_wait_asynccnt(0);
#else
  asm volatile("s_wait_asynccnt 0x0" ::: "memory");
#endif
}

// ---------------------------------------------------------------------------
// f32 -> bf16 conversion
// ---------------------------------------------------------------------------
__global__ void cvt_bf16_kernel(const float* __restrict__ src,
                                unsigned short* __restrict__ dst, int n) {
  int i = blockIdx.x * blockDim.x + threadIdx.x;
  if (i < n) dst[i] = f32_to_bf16_bits(src[i]);
}

// ---------------------------------------------------------------------------
// Generic WMMA GEMM: C[M,N] f32 = A[M,K] bf16  x  B[N,K]^T bf16
// 256 threads (8 waves), workgroup tile 128(M) x 128(N), wave tile 32x64,
// K step 32.  8 WMMA per wave per K-step against 6 fragment loads.
// ---------------------------------------------------------------------------
__global__ __launch_bounds__(256) void gemm_bf16_kernel(
    const unsigned short* __restrict__ A,
    const unsigned short* __restrict__ B,
    float* __restrict__ C, int M, int N, int K) {
  __shared__ unsigned short As[128][32];    // 8 KB
  __shared__ unsigned short Bs[128][32];    // 8 KB
  const int tid  = threadIdx.x;
  const int lane = tid & 31;
  const int wave = tid >> 5;
  const int half = lane >> 4;
  const int l16  = lane & 15;
  const int wm = (wave >> 1) * 32;          // 0,32,64,96
  const int wn = (wave & 1) * 64;           // 0,64
  const int m0 = blockIdx.y * 128;
  const int n0 = blockIdx.x * 128;

  v8f acc[2][4];
#pragma unroll
  for (int i = 0; i < 2; ++i)
#pragma unroll
    for (int j = 0; j < 4; ++j) acc[i][j] = v8f_zero();

  const int sr = tid >> 1;                  // 0..127 staging row
  const int sc = (tid & 1) * 16;            // 0 or 16 (bf16 elements)

  for (int k0 = 0; k0 < K; k0 += 32) {
    const unsigned short* ga = A + (size_t)(m0 + sr) * K + k0 + sc;
    const unsigned short* gb = B + (size_t)(n0 + sr) * K + k0 + sc;
#if ASYNC_LDS
    async_b128(ga,     &As[sr][sc]);
    async_b128(ga + 8, &As[sr][sc + 8]);
    async_b128(gb,     &Bs[sr][sc]);
    async_b128(gb + 8, &Bs[sr][sc + 8]);
    wait_async_all();
#else
    *(uint4*)(&As[sr][sc])     = *(const uint4*)ga;
    *(uint4*)(&As[sr][sc + 8]) = *(const uint4*)(ga + 8);
    *(uint4*)(&Bs[sr][sc])     = *(const uint4*)gb;
    *(uint4*)(&Bs[sr][sc + 8]) = *(const uint4*)(gb + 8);
#endif
    __syncthreads();

    if (k0 + 32 < K) {  // uniform condition: EXEC stays full
      __builtin_prefetch(A + (size_t)(m0 + sr) * K + k0 + 32 + sc, 0, 1);
      __builtin_prefetch(B + (size_t)(n0 + sr) * K + k0 + 32 + sc, 0, 1);
    }

    ABf af[2], bfr[4];
#pragma unroll
    for (int i = 0; i < 2; ++i) {
      const int row = wm + i * 16 + l16;
#pragma unroll
      for (int u = 0; u < 8; ++u) {
        const int kc = 16 * half + 2 * u;  // A-frag K pairing
        af[i].u[u] = *(const unsigned int*)(&As[row][kc]);
      }
    }
#pragma unroll
    for (int j = 0; j < 4; ++j) {
      const int col = wn + j * 16 + l16;
#pragma unroll
      for (int u = 0; u < 8; ++u)
        bfr[j].u[u] = *(const unsigned int*)(&Bs[col][16 * half + 2 * u]);
    }
#pragma unroll
    for (int i = 0; i < 2; ++i)
#pragma unroll
      for (int j = 0; j < 4; ++j)
        acc[i][j] = wmma_bf16(af[i], bfr[j], acc[i][j]);
    __syncthreads();
  }

#pragma unroll
  for (int i = 0; i < 2; ++i)
#pragma unroll
    for (int j = 0; j < 4; ++j)
#pragma unroll
      for (int r = 0; r < 8; ++r) {
        const int row = m0 + wm + i * 16 + r + 8 * half;
        const int col = n0 + wn + j * 16 + l16;
        C[(size_t)row * N + col] = acc[i][j][r];
      }
}

// ---------------------------------------------------------------------------
// Prep kernels: extract Q (+biases), K, V(transposed), r_head_k into bf16.
// ---------------------------------------------------------------------------
__global__ void prep_q_kernel(const float* __restrict__ wh,
                              const float* __restrict__ rwb,
                              const float* __restrict__ rrb,
                              unsigned short* __restrict__ qac,
                              unsigned short* __restrict__ qbd) {
  int idx = blockIdx.x * blockDim.x + threadIdx.x;
  if (idx >= BSZ * NHEAD * QLEN * DHEAD) return;
  const int d = idx & 63;
  const int i = (idx >> 6) & 1023;
  const int n = (idx >> 16) & 15;
  const int b = idx >> 20;
  const float q = wh[(size_t)((MLEN + i) * BSZ + b) * QKV_N + n * DHEAD + d];
  qac[idx] = f32_to_bf16_bits(q + rwb[n * DHEAD + d]);
  qbd[idx] = f32_to_bf16_bits(q + rrb[n * DHEAD + d]);
}

__global__ void prep_kv_kernel(const float* __restrict__ wh,
                               unsigned short* __restrict__ kbf,
                               unsigned short* __restrict__ vT) {
  int idx = blockIdx.x * blockDim.x + threadIdx.x;
  if (idx >= BSZ * NHEAD * KLEN * DHEAD) return;
  const int d = idx & 63;
  const int j = (idx >> 6) & 2047;
  const int n = (idx >> 17) & 15;
  const int b = idx >> 21;
  const size_t src = (size_t)(j * BSZ + b) * QKV_N + n * DHEAD + d;
  kbf[idx] = f32_to_bf16_bits(wh[src + NHEAD * DHEAD]);          // K block
  vT[((size_t)(b * NHEAD + n) * DHEAD + d) * KLEN + j] =
      f32_to_bf16_bits(wh[src + 2 * NHEAD * DHEAD]);             // V block, transposed
}

__global__ void prep_rk_kernel(const float* __restrict__ rh,
                               unsigned short* __restrict__ rk) {
  int idx = blockIdx.x * blockDim.x + threadIdx.x;
  if (idx >= NHEAD * KLEN * DHEAD) return;
  const int d = idx & 63;
  const int j = (idx >> 6) & 2047;
  const int n = idx >> 17;
  rk[idx] = f32_to_bf16_bits(rh[(size_t)j * DMODEL + n * DHEAD + d]);
}

// ---------------------------------------------------------------------------
// Fused flash attention with rel-shift + causal-with-memory mask.
// One wave per (b, n, 16-row query tile). 4 waves / workgroup, no cross-wave
// sync (each wave uses its own LDS scratch; same-wave LDS ops are in-order).
// ---------------------------------------------------------------------------
__global__ __launch_bounds__(128) void fa_kernel(
    const unsigned short* __restrict__ qac,
    const unsigned short* __restrict__ qbd,
    const unsigned short* __restrict__ kbf,
    const unsigned short* __restrict__ rk,
    const unsigned short* __restrict__ vT,
    unsigned short* __restrict__ av) {
  __shared__ float bdbuf[4][16][48];   // raw BD band (3 tiles) per wave
  __shared__ float sbuf[4][16][32];    // P staging for A-frag transpose
  const int tid  = threadIdx.x;
  const int wave = tid >> 5;
  const int lane = tid & 31;
  const int half = lane >> 4;
  const int l16  = lane & 15;
  const int tile = blockIdx.x * 4 + wave;       // 0..2047
  const int i0 = (tile & 63) << 4;              // query tile start
  const int n  = (tile >> 6) & (NHEAD - 1);
  const int b  = tile >> 10;

  // Preload Q fragments (AC and BD variants), 2 K-steps over DHEAD=64.
  ABf aqac[2], aqbd[2];
  {
    const size_t qrow = ((size_t)(b * NHEAD + n) * QLEN + (i0 + l16)) * DHEAD;
#pragma unroll
    for (int ks = 0; ks < 2; ++ks) {
#pragma unroll
      for (int u = 0; u < 8; ++u) {
        const int dk = 32 * ks + 16 * half + 2 * u;
        aqac[ks].u[u] = *(const unsigned int*)(qac + qrow + dk);
        aqbd[ks].u[u] = *(const unsigned int*)(qbd + qrow + dk);
      }
    }
  }

  float mrow[8], lrow[8];
  v8f oacc[4];
#pragma unroll
  for (int r = 0; r < 8; ++r) { mrow[r] = -__builtin_inff(); lrow[r] = 0.f; }
#pragma unroll
  for (int f = 0; f < 4; ++f) oacc[f] = v8f_zero();

  int jend = i0 + 16 + MLEN;        // one past last key any row in tile can see
  jend = (jend + 31) & ~31;
  if (jend > KLEN) jend = KLEN;

  const size_t kbase = (size_t)(b * NHEAD + n) * KLEN * DHEAD;
  const size_t rbase = (size_t)n * KLEN * DHEAD;
  const size_t vbase = (size_t)(b * NHEAD + n) * DHEAD * KLEN;

  for (int j0 = 0; j0 < jend; j0 += 32) {
    // ---- AC tiles: Q_ac x K^T ----
    v8f ac[2];
#pragma unroll
    for (int t = 0; t < 2; ++t) {
      v8f c = v8f_zero();
      const int col = j0 + 16 * t + l16;
#pragma unroll
      for (int ks = 0; ks < 2; ++ks) {
        ABf bf;
        const unsigned short* kp =
            kbf + kbase + (size_t)col * DHEAD + 32 * ks + 16 * half;
#pragma unroll
        for (int u = 0; u < 8; ++u)
          bf.u[u] = *(const unsigned int*)(kp + 2 * u);
        c = wmma_bf16(aqac[ks], bf, c);
      }
      ac[t] = c;
    }

    // ---- raw BD tiles covering the rel-shift band [c0, c0+47] ----
    const int c0 = j0 + (QLEN - 16) - i0;   // 16-aligned, >= 0
#pragma unroll
    for (int t2 = 0; t2 < 3; ++t2) {
      v8f c = v8f_zero();
      int col = c0 + 16 * t2 + l16;
      if (col > KLEN - 1) col = KLEN - 1;   // clamped lanes feed masked slots only
#pragma unroll
      for (int ks = 0; ks < 2; ++ks) {
        ABf bf;
        const unsigned short* rp =
            rk + rbase + (size_t)col * DHEAD + 32 * ks + 16 * half;
#pragma unroll
        for (int u = 0; u < 8; ++u)
          bf.u[u] = *(const unsigned int*)(rp + 2 * u);
        c = wmma_bf16(aqbd[ks], bf, c);
      }
#pragma unroll
      for (int r = 0; r < 8; ++r)
        bdbuf[wave][r + 8 * half][16 * t2 + l16] = c[r];
    }

    // ---- scores = (AC + shifted BD) * scale, masked ----
    float sv[2][8];
#pragma unroll
    for (int t = 0; t < 2; ++t) {
#pragma unroll
      for (int r = 0; r < 8; ++r) {
        const int M  = r + 8 * half;
        const int mi = i0 + M;
        const int nj = j0 + 16 * t + l16;
        const float bd = bdbuf[wave][M][16 * t + (l16 + 15 - M)];
        float s = (ac[t][r] + bd) * ATT_SCALE;
        if (nj > mi + MLEN) s = -__builtin_inff();
        sv[t][r] = s;
      }
    }

    // ---- online softmax update ----
    float mnew[8], alpha[8];
#pragma unroll
    for (int r = 0; r < 8; ++r) {
      float v = fmaxf(sv[0][r], sv[1][r]);
      v = fmaxf(v, __shfl_xor(v, 1, 32));
      v = fmaxf(v, __shfl_xor(v, 2, 32));
      v = fmaxf(v, __shfl_xor(v, 4, 32));
      v = fmaxf(v, __shfl_xor(v, 8, 32));
      mnew[r]  = fmaxf(mrow[r], v);
      alpha[r] = __expf(mrow[r] - mnew[r]);
      mrow[r]  = mnew[r];
    }
#pragma unroll
    for (int t = 0; t < 2; ++t)
#pragma unroll
      for (int r = 0; r < 8; ++r)
        sv[t][r] = __expf(sv[t][r] - mnew[r]);
#pragma unroll
    for (int r = 0; r < 8; ++r) {
      float s2 = sv[0][r] + sv[1][r];
      s2 += __shfl_xor(s2, 1, 32);
      s2 += __shfl_xor(s2, 2, 32);
      s2 += __shfl_xor(s2, 4, 32);
      s2 += __shfl_xor(s2, 8, 32);
      lrow[r] = lrow[r] * alpha[r] + s2;
    }
#pragma unroll
    for (int f = 0; f < 4; ++f)
#pragma unroll
      for (int r = 0; r < 8; ++r) oacc[f][r] *= alpha[r];

    // ---- stage P and rebuild it as a bf16 A-fragment (transpose via LDS) ----
#pragma unroll
    for (int t = 0; t < 2; ++t)
#pragma unroll
      for (int r = 0; r < 8; ++r)
        sbuf[wave][r + 8 * half][16 * t + l16] = sv[t][r];

    ABf pf;
#pragma unroll
    for (int u = 0; u < 8; ++u) {
      const int kc = 16 * half + 2 * u;
      const unsigned int lo = f32_to_bf16_bits(sbuf[wave][l16][kc]);
      const unsigned int hi = f32_to_bf16_bits(sbuf[wave][l16][kc + 1]);
      pf.u[u] = lo | (hi << 16);
    }

    // ---- O += P x V ----
#pragma unroll
    for (int f = 0; f < 4; ++f) {
      ABf bf;
      const unsigned short* vp =
          vT + vbase + (size_t)(f * 16 + l16) * KLEN + j0 + 16 * half;
#pragma unroll
      for (int u = 0; u < 8; ++u)
        bf.u[u] = *(const unsigned int*)(vp + 2 * u);
      oacc[f] = wmma_bf16(pf, bf, oacc[f]);
    }
  }

  // ---- normalize and write attn_vec as bf16 in (qlen, b, n*d) layout ----
#pragma unroll
  for (int f = 0; f < 4; ++f) {
#pragma unroll
    for (int r = 0; r < 8; ++r) {
      const int row = i0 + r + 8 * half;
      const int d   = f * 16 + l16;
      const float o = oacc[f][r] / lrow[r];
      av[((size_t)row * BSZ + b) * (NHEAD * DHEAD) + n * DHEAD + d] =
          f32_to_bf16_bits(o);
    }
  }
}

// ---------------------------------------------------------------------------
// Residual + LayerNorm (post-LN), one workgroup per (i,b) row.
// ---------------------------------------------------------------------------
__global__ __launch_bounds__(256) void ln_kernel(
    const float* __restrict__ w, const float* __restrict__ ao,
    const float* __restrict__ g, const float* __restrict__ bta,
    float* __restrict__ out) {
  __shared__ float red[256];
  const int row = blockIdx.x;
  const int tid = threadIdx.x;
  const float* xw = w + (size_t)row * DMODEL;
  const float* xa = ao + (size_t)row * DMODEL;

  float s = 0.f;
  for (int d = tid; d < DMODEL; d += 256) s += xw[d] + xa[d];
  red[tid] = s;
  __syncthreads();
  for (int st = 128; st > 0; st >>= 1) {
    if (tid < st) red[tid] += red[tid + st];
    __syncthreads();
  }
  const float mu = red[0] * (1.f / DMODEL);
  __syncthreads();

  float vs = 0.f;
  for (int d = tid; d < DMODEL; d += 256) {
    const float x = xw[d] + xa[d] - mu;
    vs += x * x;
  }
  red[tid] = vs;
  __syncthreads();
  for (int st = 128; st > 0; st >>= 1) {
    if (tid < st) red[tid] += red[tid + st];
    __syncthreads();
  }
  const float inv = rsqrtf(red[0] * (1.f / DMODEL) + LN_EPS);

  for (int d = tid; d < DMODEL; d += 256)
    out[(size_t)row * DMODEL + d] = (xw[d] + xa[d] - mu) * inv * g[d] + bta[d];
}

// ---------------------------------------------------------------------------
// Host-side orchestration
// ---------------------------------------------------------------------------
extern "C" void kernel_launch(void* const* d_in, const int* in_sizes, int n_in,
                              void* d_out, int out_size, void* d_ws,
                              size_t ws_size, hipStream_t stream) {
  const float* w     = (const float*)d_in[0];
  const float* r     = (const float*)d_in[1];
  const float* mems  = (const float*)d_in[2];
  // d_in[3] = attn_mask (bool) — mask is computed analytically in-kernel.
  const float* qkv_w = (const float*)d_in[4];
  const float* r_w   = (const float*)d_in[5];
  const float* o_w   = (const float*)d_in[6];
  const float* ln_g  = (const float*)d_in[7];
  const float* ln_b  = (const float*)d_in[8];
  const float* rwb   = (const float*)d_in[9];
  const float* rrb   = (const float*)d_in[10];

  char* ws = (char*)d_ws;
  size_t off = 0;
  auto take = [&](size_t bytes) -> void* {
    void* p = ws + off;
    off += (bytes + 255) & ~(size_t)255;
    return p;
  };

  unsigned short* cat_bf  = (unsigned short*)take((size_t)KLEN * BSZ * DMODEL * 2);
  unsigned short* r_bf    = (unsigned short*)take((size_t)KLEN * DMODEL * 2);
  unsigned short* qkvw_bf = (unsigned short*)take((size_t)QKV_N * DMODEL * 2);
  unsigned short* rw_bf   = (unsigned short*)take((size_t)DMODEL * DMODEL * 2);
  unsigned short* ow_bf   = (unsigned short*)take((size_t)DMODEL * DMODEL * 2);
  float* w_heads = (float*)take((size_t)KLEN * BSZ * QKV_N * 4);
  float* r_head  = (float*)take((size_t)KLEN * DMODEL * 4);
  unsigned short* qac = (unsigned short*)take((size_t)BSZ * NHEAD * QLEN * DHEAD * 2);
  unsigned short* qbd = (unsigned short*)take((size_t)BSZ * NHEAD * QLEN * DHEAD * 2);
  unsigned short* kbf = (unsigned short*)take((size_t)BSZ * NHEAD * KLEN * DHEAD * 2);
  unsigned short* vT  = (unsigned short*)take((size_t)BSZ * NHEAD * KLEN * DHEAD * 2);
  unsigned short* rk  = (unsigned short*)take((size_t)NHEAD * KLEN * DHEAD * 2);
  unsigned short* av  = (unsigned short*)take((size_t)QLEN * BSZ * DMODEL * 2);
  float* attn_out = (float*)take((size_t)QLEN * BSZ * DMODEL * 4);

  const int THR = 256;
  auto cdiv = [](int a, int b) { return (a + b - 1) / b; };

  // bf16 conversions (cat = [mems; w] is a flat concatenation in this layout)
  const int n_mem = MLEN * BSZ * DMODEL;
  const int n_wq  = QLEN * BSZ * DMODEL;
  cvt_bf16_kernel<<<cdiv(n_mem, THR), THR, 0, stream>>>(mems, cat_bf, n_mem);
  cvt_bf16_kernel<<<cdiv(n_wq, THR), THR, 0, stream>>>(w, cat_bf + n_mem, n_wq);
  cvt_bf16_kernel<<<cdiv(KLEN * DMODEL, THR), THR, 0, stream>>>(r, r_bf, KLEN * DMODEL);
  cvt_bf16_kernel<<<cdiv(QKV_N * DMODEL, THR), THR, 0, stream>>>(qkv_w, qkvw_bf, QKV_N * DMODEL);
  cvt_bf16_kernel<<<cdiv(DMODEL * DMODEL, THR), THR, 0, stream>>>(r_w, rw_bf, DMODEL * DMODEL);
  cvt_bf16_kernel<<<cdiv(DMODEL * DMODEL, THR), THR, 0, stream>>>(o_w, ow_bf, DMODEL * DMODEL);

  // projections: w_heads = cat @ qkv_w^T ; r_head = r @ r_w^T
  gemm_bf16_kernel<<<dim3(QKV_N / 128, (KLEN * BSZ) / 128), 256, 0, stream>>>(
      cat_bf, qkvw_bf, w_heads, KLEN * BSZ, QKV_N, DMODEL);
  gemm_bf16_kernel<<<dim3(DMODEL / 128, KLEN / 128), 256, 0, stream>>>(
      r_bf, rw_bf, r_head, KLEN, DMODEL, DMODEL);

  // reshapes / bias adds / transposes into WMMA-friendly bf16 layouts
  prep_q_kernel<<<cdiv(BSZ * NHEAD * QLEN * DHEAD, THR), THR, 0, stream>>>(
      w_heads, rwb, rrb, qac, qbd);
  prep_kv_kernel<<<cdiv(BSZ * NHEAD * KLEN * DHEAD, THR), THR, 0, stream>>>(
      w_heads, kbf, vT);
  prep_rk_kernel<<<cdiv(NHEAD * KLEN * DHEAD, THR), THR, 0, stream>>>(r_head, rk);

  // fused attention: AC + rel-shifted BD, mask, online softmax, P x V
  fa_kernel<<<(64 * NHEAD * BSZ) / 4, 128, 0, stream>>>(qac, qbd, kbf, rk, vT, av);

  // output projection + post-LN residual
  gemm_bf16_kernel<<<dim3(DMODEL / 128, (QLEN * BSZ) / 128), 256, 0, stream>>>(
      av, ow_bf, attn_out, QLEN * BSZ, DMODEL, DMODEL);
  ln_kernel<<<QLEN * BSZ, 256, 0, stream>>>(w, attn_out, ln_g, ln_b, (float*)d_out);

  (void)in_sizes; (void)n_in; (void)out_size; (void)ws_size;
}
